// RuntimeOptimizedTGATLayer_30820685316575
// MI455X (gfx1250) — matrix operand order repaired and checked by reading, compiled, not run on hardware
//
#include <hip/hip_runtime.h>
#include <hip/hip_bf16.h>

// ---------------------------------------------------------------------------
// TGAT layer for MI455X (gfx1250): bf16 WMMA GEMMs + 8-head flash attention.
// N=4096, IN_DIM=128, HID=512, H=8, D=64.
// HBM-critical stream = time_bias + mask (~84MB), read exactly once via
// LDS tile shared across all 8 heads (one wave per head, wave32).
// ---------------------------------------------------------------------------

#define Nn   4096
#define IND  128
#define HIDc 512
#define Hh   8
#define Dd   64
#define KB   64          // keys per attention inner iteration

#define LOG2E 1.4426950408889634f

typedef __attribute__((ext_vector_type(16))) __bf16       v16bf;
typedef __attribute__((ext_vector_type(8)))  __bf16       v8bf;
typedef __attribute__((ext_vector_type(8)))  float        v8f;
typedef __attribute__((ext_vector_type(4)))  unsigned int v4u;
typedef __attribute__((ext_vector_type(8)))  unsigned int v8u;

// Load a 16-element bf16 fragment as two contiguous 16-byte chunks.
__device__ __forceinline__ v16bf load_frag(const __bf16* p0, const __bf16* p1) {
  v4u lo = *(const v4u*)p0;
  v4u hi = *(const v4u*)p1;
  v8u c;
  c[0] = lo[0]; c[1] = lo[1]; c[2] = lo[2]; c[3] = lo[3];
  c[4] = hi[0]; c[5] = hi[1]; c[6] = hi[2]; c[7] = hi[3];
  return __builtin_bit_cast(v16bf, c);
}

__device__ __forceinline__ v8f wmma_bf16(v16bf a, v16bf b, v8f c) {
  // D = A(16x32 bf16) * B(32x16 bf16) + C(16x16 f32)
  return __builtin_amdgcn_wmma_f32_16x16x32_bf16(
      /*neg_a=*/false, a, /*neg_b=*/false, b,
      /*c_mod=*/(short)0, c, /*reuse_a=*/false, /*reuse_b=*/false);
}

// ---------------------------------------------------------------------------
// Conversion kernels (one-time, tiny traffic vs 84MB bias)
// ---------------------------------------------------------------------------
__global__ void cvt_f32_bf16(const float* __restrict__ src,
                             __bf16* __restrict__ dst, int n) {
  int i = blockIdx.x * blockDim.x + threadIdx.x;
  if (i < n) dst[i] = (__bf16)src[i];
}

// dst[n*K + k] = (bf16) src[k*Nc + n]  (weights stored transposed for WMMA-B)
__global__ void transpose_cvt(const float* __restrict__ src,
                              __bf16* __restrict__ dst, int K, int Nc) {
  int i = blockIdx.x * blockDim.x + threadIdx.x;
  if (i < K * Nc) {
    int n = i / K, kk = i % K;
    dst[i] = (__bf16)src[(size_t)kk * Nc + n];
  }
}

// ---------------------------------------------------------------------------
// Generic WMMA GEMM: Out = A[M,K](bf16,row) * Bt[Nc,K](bf16,row==B^T) + bias
// Block = 128 threads = 4 waves; each wave owns one 16x16 output tile.
// All intermediates (~24MB) stay L2-resident (192MB), so the redundant
// A-tile reads across the 8 column-blocks are L2 hits, not HBM traffic.
// ---------------------------------------------------------------------------
enum { OUT_BF16_ROW = 0, OUT_BF16_TRANS = 1, OUT_F32_ROW = 2 };

template <int MODE>
__global__ __launch_bounds__(128)
void gemm_wmma(const __bf16* __restrict__ A, const __bf16* __restrict__ Bt,
               const float* __restrict__ bias, void* __restrict__ Out,
               int M, int K, int Nc) {
  const int lane = threadIdx.x & 31;
  const int wave = threadIdx.x >> 5;
  const int lc   = lane & 15;
  const int half = lane >> 4;
  const int m_base = blockIdx.x * 16;
  const int n      = blockIdx.y * 64 + wave * 16 + lc;

  const __bf16* arow = A  + (size_t)(m_base + lc) * K;
  const __bf16* brow = Bt + (size_t)n * K;

  v8f c = {};
  for (int kb = 0; kb < K; kb += 32) {
    // A frag: K = half*16..+16 (contiguous, matches B)
    v16bf af = load_frag(arow + kb + half * 16, arow + kb + half * 16 + 8);
    // B frag: lane = column n ; K = half*16..+16 (contiguous in Bt row)
    v16bf bf_ = load_frag(brow + kb + half * 16, brow + kb + half * 16 + 8);
    c = wmma_bf16(af, bf_, c);
  }

  const float bn = bias[n];
  if constexpr (MODE == OUT_BF16_ROW) {
    __bf16* o = (__bf16*)Out;
#pragma unroll
    for (int r = 0; r < 8; ++r)
      o[(size_t)(m_base + r + 8 * half) * Nc + n] = (__bf16)(c[r] + bn);
  } else if constexpr (MODE == OUT_BF16_TRANS) {
    // Out[n*M + m]: 8 consecutive rows per lane -> one 16B store
    __bf16* o = (__bf16*)Out;
    v8bf pk;
#pragma unroll
    for (int r = 0; r < 8; ++r) pk[r] = (__bf16)(c[r] + bn);
    *(v4u*)(o + (size_t)n * M + m_base + 8 * half) = __builtin_bit_cast(v4u, pk);
  } else {
    float* o = (float*)Out;
#pragma unroll
    for (int r = 0; r < 8; ++r)
      o[(size_t)(m_base + r + 8 * half) * Nc + n] = c[r] + bn;
  }
}

// ---------------------------------------------------------------------------
// Flash attention: one block = one 16-query tile, 8 waves = 8 heads.
// - 64-key inner blocks: 16 WMMAs per online-softmax pass (halves the
//   shuffle/rescale VALU overhead per FLOP vs 32-key blocks).
// - Ping-pong bias LDS buffer: one barrier per iteration; the next bias
//   tile's global loads overlap WMMA compute.
// - log2-domain softmax: bias pre-scaled by log2(e) in LDS, scores use a
//   single FMA with 0.125*log2(e); exponentials are bare v_exp_f32.
// ---------------------------------------------------------------------------
__global__ __launch_bounds__(256)
void attn_flash(const __bf16* __restrict__ q, const __bf16* __restrict__ k,
                const __bf16* __restrict__ vt,             // vt[HID][N]
                const unsigned char* __restrict__ mask,    // [N,N] bool
                const float* __restrict__ tb,              // [N,N] f32
                __bf16* __restrict__ ao) {                 // [N,HID]
  __shared__ float  lds_bias[2][16 * KB];       // ping-pong, log2-scaled
  __shared__ __bf16 p_stage[8 * 16 * KB];       // per-wave probs staging

  const int tid  = threadIdx.x;
  const int lane = tid & 31;
  const int h    = tid >> 5;          // wave id == head id
  const int lc   = lane & 15;
  const int half = lane >> 4;
  const int qb   = blockIdx.x * 16;
  const float SC = 0.125f * LOG2E;    // 1/sqrt(64) * log2(e)

  // Q fragments for this head (reused across the whole key loop)
  const __bf16* qrow = q + (size_t)(qb + lc) * HIDc + h * Dd;
  const v16bf aq0 = load_frag(qrow + half * 16,      qrow + half * 16 + 8);
  const v16bf aq1 = load_frag(qrow + 32 + half * 16, qrow + 40 + half * 16);

  float m_r[8], l_r[8];
  v8f O[4] = {};
#pragma unroll
  for (int r = 0; r < 8; ++r) { m_r[r] = -3.0e38f; l_r[r] = 0.f; }

  __bf16* st = p_stage + h * (16 * KB);

  // ---- cooperative bias tile fill: 1024 f32, 4 per thread, vectorized ----
  auto fill_bias = [&](int kbn, int buf) {
    const int e   = tid * 4;                    // 256 threads * 4 = 1024
    const int row = e >> 6, col = e & 63;
    const size_t idx = (size_t)(qb + row) * Nn + kbn + col;
    const float4 t  = *(const float4*)(tb + idx);
    const uchar4 mk = *(const uchar4*)(mask + idx);
    float* d = &lds_bias[buf][e];
    d[0] = (t.x + (mk.x ? -1.0e9f : 0.0f)) * LOG2E;
    d[1] = (t.y + (mk.y ? -1.0e9f : 0.0f)) * LOG2E;
    d[2] = (t.z + (mk.z ? -1.0e9f : 0.0f)) * LOG2E;
    d[3] = (t.w + (mk.w ? -1.0e9f : 0.0f)) * LOG2E;
  };

  fill_bias(0, 0);
  int cur = 0;

  for (int kb = 0; kb < Nn; kb += KB) {
    __syncthreads();  // bias[cur] ready; bias[cur^1] free for prefetch
    if (kb + KB < Nn) fill_bias(kb + KB, cur ^ 1);
    const float* biasL = lds_bias[cur];

    // ---- scores: four 16x16 tiles over 64 keys, K-dim = 64 head dims ----
    v8f s[4];
#pragma unroll
    for (int j = 0; j < 4; ++j) {
      const __bf16* kr = k + (size_t)(kb + j * 16 + lc) * HIDc + h * Dd + half * 16;
      v8f cc = {};
      cc = wmma_bf16(aq0, load_frag(kr, kr + 8), cc);
      cc = wmma_bf16(aq1, load_frag(kr + 32, kr + 40), cc);
      s[j] = cc;
    }

    // ---- online softmax in log2 domain (per C-layout row r+8*half) ----
#pragma unroll
    for (int r = 0; r < 8; ++r) {
      const int mrow = (r + 8 * half) * KB;
      float a[4];
#pragma unroll
      for (int j = 0; j < 4; ++j)
        a[j] = s[j][r] * SC + biasL[mrow + j * 16 + lc];
      float mx = fmaxf(fmaxf(a[0], a[1]), fmaxf(a[2], a[3]));
#pragma unroll
      for (int d = 1; d < 16; d <<= 1) mx = fmaxf(mx, __shfl_xor(mx, d, 32));
      const float mnew = fmaxf(m_r[r], mx);
      const float corr = exp2f(m_r[r] - mnew);
      float p[4], ps = 0.f;
#pragma unroll
      for (int j = 0; j < 4; ++j) { p[j] = exp2f(a[j] - mnew); ps += p[j]; }
#pragma unroll
      for (int d = 1; d < 16; d <<= 1) ps += __shfl_xor(ps, d, 32);
      l_r[r] = l_r[r] * corr + ps;
      m_r[r] = mnew;
#pragma unroll
      for (int dt = 0; dt < 4; ++dt) O[dt][r] *= corr;
#pragma unroll
      for (int j = 0; j < 4; ++j)
        st[mrow + j * 16 + lc] = (__bf16)p[j];
    }
    // per-wave LDS staging: DS ops are in-order within a wave, no barrier

    // ---- probs C-layout -> A-layout: two 16x32 A fragments ----
    const __bf16* sp = st + lc * KB;
    const v16bf ap0 = load_frag(sp + half * 16,      sp + half * 16 + 8);
    const v16bf ap1 = load_frag(sp + 32 + half * 16, sp + 40 + half * 16);

    // ---- O += P(16x64) * V(64keys x 64dims): 8 WMMAs ----
#pragma unroll
    for (int dt = 0; dt < 4; ++dt) {
      const __bf16* vr = vt + (size_t)(h * Dd + dt * 16 + lc) * Nn + kb + half * 16;
      O[dt] = wmma_bf16(ap0, load_frag(vr, vr + 8), O[dt]);
      O[dt] = wmma_bf16(ap1, load_frag(vr + 32, vr + 40), O[dt]);
    }
    cur ^= 1;
  }

  // ---- finalize: divide by l, store bf16 row-major [N,HID] ----
#pragma unroll
  for (int r = 0; r < 8; ++r) {
    const float inv = 1.0f / fmaxf(l_r[r], 1e-30f);
    const int m = qb + r + 8 * half;
#pragma unroll
    for (int dt = 0; dt < 4; ++dt)
      ao[(size_t)m * HIDc + h * Dd + dt * 16 + lc] = (__bf16)(O[dt][r] * inv);
  }
}

// ---------------------------------------------------------------------------
// Launcher
// ---------------------------------------------------------------------------
extern "C" void kernel_launch(void* const* d_in, const int* in_sizes, int n_in,
                              void* d_out, int out_size, void* d_ws, size_t ws_size,
                              hipStream_t stream) {
  (void)in_sizes; (void)n_in; (void)out_size; (void)ws_size;

  const float*         nf   = (const float*)d_in[0];
  const unsigned char* mask = (const unsigned char*)d_in[1];
  const float*         tb   = (const float*)d_in[2];
  const float* W_in = (const float*)d_in[3];  const float* b_in = (const float*)d_in[4];
  const float* Wq   = (const float*)d_in[5];  const float* bq   = (const float*)d_in[6];
  const float* Wk   = (const float*)d_in[7];  const float* bk   = (const float*)d_in[8];
  const float* Wv   = (const float*)d_in[9];  const float* bv   = (const float*)d_in[10];
  const float* Wo   = (const float*)d_in[11]; const float* bo   = (const float*)d_in[12];
  float* out = (float*)d_out;

  // Workspace carve-up (~23.2 MB total; intermediates all L2-resident)
  char* ws = (char*)d_ws;
  __bf16* nf_b  = (__bf16*)ws; ws += (size_t)Nn * IND * 2;
  __bf16* Wint  = (__bf16*)ws; ws += (size_t)HIDc * IND * 2;
  __bf16* Wqt   = (__bf16*)ws; ws += (size_t)HIDc * HIDc * 2;
  __bf16* Wkt   = (__bf16*)ws; ws += (size_t)HIDc * HIDc * 2;
  __bf16* Wvt   = (__bf16*)ws; ws += (size_t)HIDc * HIDc * 2;
  __bf16* Wot   = (__bf16*)ws; ws += (size_t)HIDc * HIDc * 2;
  __bf16* x_b   = (__bf16*)ws; ws += (size_t)Nn * HIDc * 2;
  __bf16* q_b   = (__bf16*)ws; ws += (size_t)Nn * HIDc * 2;
  __bf16* k_b   = (__bf16*)ws; ws += (size_t)Nn * HIDc * 2;
  __bf16* vt_b  = (__bf16*)ws; ws += (size_t)Nn * HIDc * 2;   // [HID][N]
  __bf16* ao_b  = (__bf16*)ws; ws += (size_t)Nn * HIDc * 2;

  const int T = 256;
  cvt_f32_bf16<<<(Nn * IND + T - 1) / T, T, 0, stream>>>(nf, nf_b, Nn * IND);
  transpose_cvt<<<(IND * HIDc + T - 1) / T, T, 0, stream>>>(W_in, Wint, IND, HIDc);
  transpose_cvt<<<(HIDc * HIDc + T - 1) / T, T, 0, stream>>>(Wq, Wqt, HIDc, HIDc);
  transpose_cvt<<<(HIDc * HIDc + T - 1) / T, T, 0, stream>>>(Wk, Wkt, HIDc, HIDc);
  transpose_cvt<<<(HIDc * HIDc + T - 1) / T, T, 0, stream>>>(Wv, Wvt, HIDc, HIDc);
  transpose_cvt<<<(HIDc * HIDc + T - 1) / T, T, 0, stream>>>(Wo, Wot, HIDc, HIDc);

  dim3 gg(Nn / 16, HIDc / 64);   // 256 x 8
  gemm_wmma<OUT_BF16_ROW>  <<<gg, 128, 0, stream>>>(nf_b, Wint, b_in, x_b,  Nn, IND,  HIDc);
  gemm_wmma<OUT_BF16_ROW>  <<<gg, 128, 0, stream>>>(x_b,  Wqt,  bq,   q_b,  Nn, HIDc, HIDc);
  gemm_wmma<OUT_BF16_ROW>  <<<gg, 128, 0, stream>>>(x_b,  Wkt,  bk,   k_b,  Nn, HIDc, HIDc);
  gemm_wmma<OUT_BF16_TRANS><<<gg, 128, 0, stream>>>(x_b,  Wvt,  bv,   vt_b, Nn, HIDc, HIDc);

  attn_flash<<<Nn / 16, 256, 0, stream>>>(q_b, k_b, vt_b, mask, tb, ao_b);

  gemm_wmma<OUT_F32_ROW>   <<<gg, 128, 0, stream>>>(ao_b, Wot,  bo,   out,  Nn, HIDc, HIDc);
}
